// SegAwareMultiHeadAttention_206158430720
// MI455X (gfx1250) — matrix operand order repaired and checked by reading, not compiled, not run on hardware
//
#include <hip/hip_runtime.h>

#define HIDDEN 768
#define HEADS  12
#define HDIM   64
#define BATCH  32
#define SEQ    577
#define SEQPAD 640               // padded to multiple of 64 (query/key tile)
#define MROWS  (BATCH * SEQ)     // 18464 (not divisible by 64)
#define QKVN   (3 * HIDDEN)      // 2304

typedef __bf16 v16bf __attribute__((ext_vector_type(16)));
typedef float  v8f   __attribute__((ext_vector_type(8)));
typedef int    v4i   __attribute__((ext_vector_type(4)));

union Frag {
  v16bf v;
  unsigned short e[16];
  unsigned int   u[8];
  uint4          q[2];
};

// ---- async global->LDS path (CDNA5), feature-guarded -----------------------
#if defined(__has_builtin)
#if __has_builtin(__builtin_amdgcn_global_load_async_to_lds_b128) && \
    __has_builtin(__builtin_amdgcn_s_wait_asynccnt)
#define ASYNC_LDS 1
#endif
#endif
#ifndef ASYNC_LDS
#define ASYNC_LDS 0
#endif

__device__ __forceinline__ void lds_fill_b128(void* lds, const void* g) {
#if ASYNC_LDS
  // AS3 pointer from integer: low 32 bits of the flat address are the LDS
  // offset.
  __builtin_amdgcn_global_load_async_to_lds_b128(
      (__attribute__((address_space(1))) v4i*)(size_t)g,
      (__attribute__((address_space(3))) v4i*)(size_t)lds, 0, 0);
#else
  *(uint4*)lds = *(const uint4*)g;
#endif
}

__device__ __forceinline__ void async_wait0() {
#if ASYNC_LDS
  __builtin_amdgcn_s_wait_asynccnt(0);
#endif
}
// ---------------------------------------------------------------------------

// fp32 -> bf16 round-to-nearest-even, pure integer path
__device__ __forceinline__ unsigned short f2bf(float f) {
  unsigned int x = __builtin_bit_cast(unsigned int, f);
  x += 0x7fffu + ((x >> 16) & 1u);
  return (unsigned short)(x >> 16);
}

__device__ __forceinline__ v8f wmma_bf16(const Frag& a, const Frag& b, v8f c) {
  return __builtin_amdgcn_wmma_f32_16x16x32_bf16(false, a.v, false, b.v,
                                                 (short)0, c, false, false);
}

__global__ void cvt_f32_to_bf16(const float* __restrict__ src,
                                unsigned short* __restrict__ dst, int n) {
  int i = blockIdx.x * blockDim.x + threadIdx.x;
  if (i < n) dst[i] = f2bf(src[i]);
}

// W (K x N fp32, row major) -> Wt (N x K bf16, row major); one-time, tiny
__global__ void cvt_transpose_bf16(const float* __restrict__ src,
                                   unsigned short* __restrict__ dst,
                                   int K, int N) {
  const int n = blockIdx.x * blockDim.x + threadIdx.x;
  const int k = blockIdx.y;
  if (n < N) dst[(size_t)n * K + k] = f2bf(src[(size_t)k * N + n]);
}

__global__ void zero_u32(unsigned int* __restrict__ p, int n) {
  for (int i = blockIdx.x * blockDim.x + threadIdx.x; i < n;
       i += gridDim.x * blockDim.x)
    p[i] = 0u;
}

// ---------------------------------------------------------------------------
// 64x64 output tile GEMM: C = A (M x 768, bf16) . Wt^T (+bias), Wt is N x 768.
// 128 threads = 4 waves; wave w owns rows [w*16, w*16+16), all 64 cols.
// Double-buffered LDS K-slabs filled with async global->LDS b128.
// MODE 0: QKV — scatter into padded q/k (B,H,SEQPAD,HDIM) and vT (B,H,HDIM,SEQPAD).
// MODE 1: proj — write fp32 to out (row-major M x 768).
// ---------------------------------------------------------------------------
template <int MODE>
__global__ __launch_bounds__(128) void gemm64x64(
    const unsigned short* __restrict__ A,    // M x 768 bf16 (row major)
    const unsigned short* __restrict__ Wt,   // Nw x 768 bf16 (row major, W^T)
    const float* __restrict__ bias,          // Nw
    unsigned short* __restrict__ qb,
    unsigned short* __restrict__ kb,
    unsigned short* __restrict__ vbT,
    float* __restrict__ out) {
  __shared__ unsigned short As[2][64 * 32];  // [m][k]
  __shared__ unsigned short Bs[2][64 * 32];  // [n][k]  (from Wt rows)

  const int tid  = threadIdx.x;
  const int wave = tid >> 5;
  const int lane = tid & 31;
  const int hi   = lane >> 4;
  const int ln   = lane & 15;
  const int rowBase = blockIdx.x * 64;
  const int colBase = blockIdx.y * 64;

  auto loadTiles = [&](int k0, int buf) {
#pragma unroll
    for (int t = 0; t < 2; ++t) {
      const int u = tid * 2 + t;
      const int r = u >> 2, c = u & 3;
      lds_fill_b128(&As[buf][u * 8],
                    &A[(size_t)(rowBase + r) * 768 + k0 + c * 8]);
      lds_fill_b128(&Bs[buf][u * 8],
                    &Wt[(size_t)(colBase + r) * 768 + k0 + c * 8]);
    }
  };

  v8f acc[4] = {};

  loadTiles(0, 0);
  async_wait0();
  __syncthreads();

  for (int k0 = 0; k0 < 768; k0 += 32) {
    const int buf = (k0 >> 5) & 1;
    if (k0 + 32 < 768) loadTiles(k0 + 32, buf ^ 1);  // DMA next slab now

    // A fragment: 2 x b128
    Frag a;
    const int m = wave * 16 + ln;
    a.q[0] = *(const uint4*)&As[buf][m * 32 + hi * 8];
    a.q[1] = *(const uint4*)&As[buf][m * 32 + 16 + hi * 8];
#pragma unroll
    for (int j = 0; j < 4; ++j) {
      Frag bf;
      const int n = j * 16 + ln;
      bf.q[0] = *(const uint4*)&Bs[buf][n * 32 + hi * 8];
      bf.q[1] = *(const uint4*)&Bs[buf][n * 32 + 16 + hi * 8];
      acc[j] = wmma_bf16(a, bf, acc[j]);
    }

    async_wait0();   // next slab landed
    __syncthreads(); // everyone done with current slab
  }

#pragma unroll
  for (int j = 0; j < 4; ++j) {
#pragma unroll
    for (int i = 0; i < 8; ++i) {
      const int rg = rowBase + wave * 16 + i + hi * 8;  // < MROWS always
      const int cg = colBase + j * 16 + ln;
      const float val = acc[j][i] + bias[cg];
      if (MODE == 0) {
        const int s   = cg / HIDDEN;
        const int rem = cg - s * HIDDEN;
        const int hh  = rem >> 6;
        const int d   = rem & 63;
        const int bb  = rg / SEQ;
        const int n   = rg - bb * SEQ;
        const unsigned short bv = f2bf(val);
        if (s == 0)
          qb[((size_t)(bb * HEADS + hh) * SEQPAD + n) * HDIM + d] = bv;
        else if (s == 1)
          kb[((size_t)(bb * HEADS + hh) * SEQPAD + n) * HDIM + d] = bv;
        else  // V stored transposed: (B,H,HDIM,SEQPAD)
          vbT[((size_t)(bb * HEADS + hh) * HDIM + d) * SEQPAD + n] = bv;
      } else {
        out[(size_t)rg * HIDDEN + cg] = val;
      }
    }
  }
}

// ---------------------------------------------------------------------------
// Flash attention: block = 4 waves, 64 query rows per block, one (b,h).
// Online softmax; CLS-row segment bias; mask padded keys (>= SEQ).
// Q/K rows are (SEQPAD,HDIM); V is transposed (HDIM,SEQPAD).
// Double-buffered K/V tiles filled with async global->LDS b128.
// ---------------------------------------------------------------------------
__global__ __launch_bounds__(128) void attn_flash(
    const unsigned short* __restrict__ qb,
    const unsigned short* __restrict__ kb,
    const unsigned short* __restrict__ vbT,
    const float* __restrict__ seg_bias,   // (B, SEQ-1)
    const float* __restrict__ seg_scale,  // scalar
    unsigned short* __restrict__ attn_out /* (B*SEQ, HIDDEN) bf16 */) {
  __shared__ unsigned short Kt[2][64 * 64];  // [key][dim]
  __shared__ unsigned short Vt[2][64 * 64];  // [dim][key]
  __shared__ unsigned short Pt[4][16 * 64];  // per-wave [row][key]

  const int tid  = threadIdx.x;
  const int wave = tid >> 5;
  const int lane = tid & 31;
  const int hi   = lane >> 4;
  const int ln   = lane & 15;
  const int qt = blockIdx.x;
  const int h  = blockIdx.y;
  const int b  = blockIdx.z;
  const int bh = b * HEADS + h;
  const size_t baseQK = (size_t)bh * SEQPAD * HDIM;  // q/k base (elements)
  const size_t baseV  = (size_t)bh * HDIM * SEQPAD;  // vT base (elements)
  const float sscale = seg_scale[0];
  const float scale  = 0.125f;  // 1/sqrt(64)

  auto loadKV = [&](int kt, int buf) {
#pragma unroll
    for (int t = 0; t < 4; ++t) {
      const int u = tid * 4 + t;
      const int r = u >> 3, c = u & 7;
      lds_fill_b128(&Kt[buf][u * 8],
                    &kb[baseQK + (size_t)(kt + r) * HDIM + c * 8]);
      lds_fill_b128(&Vt[buf][u * 8],
                    &vbT[baseV + (size_t)r * SEQPAD + kt + c * 8]);
    }
  };

  // Q A-fragments, resident for the whole key loop: 2 x global b128 each
  Frag qa[2];
  {
    const size_t qoff = baseQK + (size_t)(qt * 64 + wave * 16 + ln) * HDIM;
#pragma unroll
    for (int f = 0; f < 2; ++f) {
      qa[f].q[0] = *(const uint4*)&qb[qoff + f * 32 + hi * 8];
      qa[f].q[1] = *(const uint4*)&qb[qoff + f * 32 + 16 + hi * 8];
    }
  }

  v8f o[4] = {};
  float mrow[8], lrow[8];
#pragma unroll
  for (int i = 0; i < 8; ++i) { mrow[i] = -1e30f; lrow[i] = 0.0f; }

  const int nrow0 = qt * 64 + wave * 16 + hi * 8;  // lane's rows: nrow0 + i

  loadKV(0, 0);
  async_wait0();
  __syncthreads();

  for (int kt = 0; kt < SEQPAD; kt += 64) {
    const int buf = (kt >> 6) & 1;
    if (kt + 64 < SEQPAD) loadKV(kt + 64, buf ^ 1);  // DMA next tile now

    // S = Q . K^T  (4 key blocks of 16, K-dim 64 in two wmma)
    v8f s[4];
#pragma unroll
    for (int j = 0; j < 4; ++j) {
      v8f acc = {};
#pragma unroll
      for (int f = 0; f < 2; ++f) {
        Frag kf;  // B[kdim][key] = Kt[key][kdim]: pair-contiguous along dim
        const int key = j * 16 + ln;
        kf.q[0] = *(const uint4*)&Kt[buf][key * 64 + f * 32 + hi * 8];
        kf.q[1] = *(const uint4*)&Kt[buf][key * 64 + f * 32 + 16 + hi * 8];
        acc = wmma_bf16(qa[f], kf, acc);
      }
      s[j] = acc;
    }

    // scale + seg bias + mask, then online softmax (rows live across 16 lanes)
#pragma unroll
    for (int i = 0; i < 8; ++i) {
      float rmax = -1e30f;
#pragma unroll
      for (int j = 0; j < 4; ++j) {
        const int keyg = kt + j * 16 + ln;
        float val = s[j][i] * scale;
        if ((nrow0 + i) == 0 && keyg >= 1 && keyg < SEQ)
          val += seg_bias[b * (SEQ - 1) + keyg - 1] * sscale;
        if (keyg >= SEQ) val = -1e30f;
        s[j][i] = val;
        rmax = fmaxf(rmax, val);
      }
#pragma unroll
      for (int msk = 1; msk <= 8; msk <<= 1)
        rmax = fmaxf(rmax, __shfl_xor(rmax, msk, 32));
      const float mn = fmaxf(mrow[i], rmax);
      const float fs = __expf(mrow[i] - mn);
      mrow[i] = mn;
      float ps = 0.0f;
#pragma unroll
      for (int j = 0; j < 4; ++j) {
        const float p = __expf(s[j][i] - mn);
        s[j][i] = p;
        ps += p;
      }
#pragma unroll
      for (int msk = 1; msk <= 8; msk <<= 1)
        ps += __shfl_xor(ps, msk, 32);
      lrow[i] = lrow[i] * fs + ps;
#pragma unroll
      for (int j = 0; j < 4; ++j) o[j][i] *= fs;
    }

    // C-layout P -> per-wave LDS -> A-layout fragments (same-wave LDS in-order)
#pragma unroll
    for (int j = 0; j < 4; ++j)
#pragma unroll
      for (int i = 0; i < 8; ++i)
        Pt[wave][(i + hi * 8) * 64 + j * 16 + ln] = f2bf(s[j][i]);

    // O += P . V
#pragma unroll
    for (int f = 0; f < 2; ++f) {
      Frag pa;  // A-layout, pair-contiguous
      pa.q[0] = *(const uint4*)&Pt[wave][ln * 64 + f * 32 + hi * 8];
      pa.q[1] = *(const uint4*)&Pt[wave][ln * 64 + f * 32 + 16 + hi * 8];
#pragma unroll
      for (int j = 0; j < 4; ++j) {
        Frag vf;  // B[key][dim] = Vt[dim][key]: pair-contiguous along key
        const int nd = j * 16 + ln;
        vf.q[0] = *(const uint4*)&Vt[buf][nd * 64 + f * 32 + hi * 8];
        vf.q[1] = *(const uint4*)&Vt[buf][nd * 64 + f * 32 + 16 + hi * 8];
        o[j] = wmma_bf16(pa, vf, o[j]);
      }
    }

    async_wait0();   // next K/V tile landed
    __syncthreads(); // everyone done with current tile
  }

  // normalize + write (B*SEQ, HIDDEN) bf16 for the projection GEMM
#pragma unroll
  for (int j = 0; j < 4; ++j)
#pragma unroll
    for (int i = 0; i < 8; ++i) {
      const int n = qt * 64 + wave * 16 + i + hi * 8;
      if (n < SEQ) {
        const float val = o[j][i] / lrow[i];
        attn_out[((size_t)(b * SEQ + n)) * HIDDEN + h * HDIM + j * 16 + ln] =
            f2bf(val);
      }
    }
}

extern "C" void kernel_launch(void* const* d_in, const int* in_sizes, int n_in,
                              void* d_out, int out_size, void* d_ws,
                              size_t ws_size, hipStream_t stream) {
  (void)in_sizes; (void)n_in; (void)out_size; (void)ws_size;
  const float* x         = (const float*)d_in[0];
  const float* seg_bias  = (const float*)d_in[1];
  const float* w_qkv     = (const float*)d_in[2];
  const float* b_qkv     = (const float*)d_in[3];
  const float* w_proj    = (const float*)d_in[4];
  const float* b_proj    = (const float*)d_in[5];
  const float* seg_scale = (const float*)d_in[6];
  float* out = (float*)d_out;

  char* ws = (char*)d_ws;
  const size_t SZ_XBF     = (size_t)MROWS * HIDDEN * 2;                 // 28.4 MB
  const size_t SZ_WQKV    = (size_t)HIDDEN * QKVN * 2;                  // 3.5 MB
  const size_t SZ_WPROJ   = (size_t)HIDDEN * HIDDEN * 2;                // 1.2 MB
  const size_t SZ_HEADBUF = (size_t)BATCH * HEADS * SEQPAD * HDIM * 2;  // 31.5 MB

  unsigned short* xbf     = (unsigned short*)(ws);
  unsigned short* wqkvT   = (unsigned short*)(ws + SZ_XBF);
  unsigned short* wprojT  = (unsigned short*)(ws + SZ_XBF + SZ_WQKV);
  unsigned short* qbb     = (unsigned short*)(ws + SZ_XBF + SZ_WQKV + SZ_WPROJ);
  unsigned short* kbb     = qbb + SZ_HEADBUF / 2;
  unsigned short* vbbT    = kbb + SZ_HEADBUF / 2;
  unsigned short* attnbf  = (unsigned short*)((char*)qbb + 3 * SZ_HEADBUF);

  const int nx = MROWS * HIDDEN;
  cvt_f32_to_bf16<<<(nx + 255) / 256, 256, 0, stream>>>(x, xbf, nx);
  // one-time weight transposes (W is K x N; store N x K bf16)
  cvt_transpose_bf16<<<dim3((QKVN + 255) / 256, HIDDEN), 256, 0, stream>>>(
      w_qkv, wqkvT, HIDDEN, QKVN);
  cvt_transpose_bf16<<<dim3((HIDDEN + 255) / 256, HIDDEN), 256, 0, stream>>>(
      w_proj, wprojT, HIDDEN, HIDDEN);

  // zero q/k/v so padded rows/cols [SEQ, SEQPAD) are clean
  const int nz = (int)(3 * SZ_HEADBUF / 4);
  zero_u32<<<2048, 256, 0, stream>>>((unsigned int*)qbb, nz);

  gemm64x64<0><<<dim3(MROWS / 64, QKVN / 64), 128, 0, stream>>>(
      xbf, wqkvT, b_qkv, qbb, kbb, vbbT, nullptr);

  attn_flash<<<dim3(SEQPAD / 64, HEADS, BATCH), 128, 0, stream>>>(
      qbb, kbb, vbbT, seg_bias, seg_scale, attnbf);

  gemm64x64<1><<<dim3(MROWS / 64, HIDDEN / 64), 128, 0, stream>>>(
      attnbf, wprojT, b_proj, nullptr, nullptr, nullptr, out);
}
